// TimeMix_79903571574985
// MI455X (gfx1250) — matrix operand order repaired and checked by reading, not compiled, not run on hardware
//
#include <hip/hip_runtime.h>

// ---------------------------------------------------------------------------
// Problem constants (match reference setup_inputs)
// ---------------------------------------------------------------------------
#define T_DIM 8192
#define C_DIM 2048

// GEMM tiling: 128x128 block tile, K-step 64, 8 waves (4x2), wave tile 32x64
#define BM 128
#define BN 128
#define BK 64

typedef __attribute__((ext_vector_type(16))) __bf16          v16bf;
typedef __attribute__((ext_vector_type(8)))  float           v8f;
typedef __attribute__((ext_vector_type(16))) unsigned short  v16u;
typedef __attribute__((ext_vector_type(8)))  unsigned short  v8u;
typedef int v4i_vs __attribute__((__vector_size__(4 * sizeof(int))));

#define AS1 __attribute__((address_space(1)))
#define AS3 __attribute__((address_space(3)))

#if defined(__has_builtin)
#if __has_builtin(__builtin_amdgcn_global_load_async_to_lds_b128) && \
    __has_builtin(__builtin_amdgcn_s_wait_asynccnt)
#define USE_ASYNC_LDS 1
#endif
#endif
#ifndef USE_ASYNC_LDS
#define USE_ASYNC_LDS 0
#endif

__device__ __forceinline__ unsigned short f2bf(float f) {
    unsigned int u = __builtin_bit_cast(unsigned int, f);
    unsigned int r = u + 0x7fffu + ((u >> 16) & 1u);   // round-to-nearest-even
    return (unsigned short)(r >> 16);
}

// ---------------------------------------------------------------------------
// 1) Weight transpose + f32 -> bf16   (Wt[n][k] = bf16(W[k][n]))
// ---------------------------------------------------------------------------
__global__ __launch_bounds__(256)
void transpose_bf16_kernel(const float* __restrict__ W, unsigned short* __restrict__ Wt) {
    __shared__ float tile[32][33];
    const int kb = blockIdx.x * 32;     // input row (k) base
    const int nb = blockIdx.y * 32;     // input col (n) base
    const int tx = threadIdx.x;         // 0..31
    const int ty = threadIdx.y;         // 0..7
    #pragma unroll
    for (int i = 0; i < 32; i += 8)
        tile[ty + i][tx] = W[(size_t)(kb + ty + i) * C_DIM + (nb + tx)];
    __syncthreads();
    #pragma unroll
    for (int i = 0; i < 32; i += 8)
        Wt[(size_t)(nb + ty + i) * C_DIM + (kb + tx)] = f2bf(tile[tx][ty + i]);
}

// ---------------------------------------------------------------------------
// 2) Fused LayerNorm + token-shift mixing.  One block per timestep t.
// ---------------------------------------------------------------------------
__device__ __forceinline__ float block_sum(float v, float* red, int tid) {
    red[tid] = v; __syncthreads();
    #pragma unroll
    for (int s = 128; s > 0; s >>= 1) {
        if (tid < s) red[tid] += red[tid + s];
        __syncthreads();
    }
    float r = red[0]; __syncthreads();
    return r;
}

__global__ __launch_bounds__(256)
void ln_mix_kernel(const float* __restrict__ x,   const float* __restrict__ sx,
                   const float* __restrict__ tmk, const float* __restrict__ tmv,
                   const float* __restrict__ tmr,
                   const float* __restrict__ lns, const float* __restrict__ lnb,
                   unsigned short* __restrict__ Ak, unsigned short* __restrict__ Av,
                   unsigned short* __restrict__ Ar, float* __restrict__ xx_last) {
    __shared__ float red[256];
    const int t   = blockIdx.x;
    const int tid = threadIdx.x;

    const float* xt = x + (size_t)t * C_DIM;
    const float* xp = x + (size_t)(t > 0 ? t - 1 : 0) * C_DIM;

    float xv[8], pv[8];
    float s0 = 0.f, q0 = 0.f, s1 = 0.f, q1 = 0.f;
    #pragma unroll
    for (int i = 0; i < 8; i++) {
        const int c = tid + i * 256;
        float a = xt[c]; xv[i] = a; s0 += a; q0 += a * a;
        float b = xp[c]; pv[i] = b; s1 += b; q1 += b * b;
    }
    const float S0 = block_sum(s0, red, tid);
    const float Q0 = block_sum(q0, red, tid);
    const float S1 = block_sum(s1, red, tid);
    const float Q1 = block_sum(q1, red, tid);

    const float invC = 1.0f / (float)C_DIM;
    const float mu0 = S0 * invC;
    const float rs0 = __frsqrt_rn(fmaxf(Q0 * invC - mu0 * mu0, 0.f) + 1e-5f);
    const float mu1 = S1 * invC;
    const float rs1 = __frsqrt_rn(fmaxf(Q1 * invC - mu1 * mu1, 0.f) + 1e-5f);

    #pragma unroll
    for (int i = 0; i < 8; i++) {
        const int c = tid + i * 256;
        const float g = lns[c], bta = lnb[c];
        const float xx  = (xv[i] - mu0) * rs0 * g + bta;
        const float sxs = (t == 0) ? sx[c] : (pv[i] - mu1) * rs1 * g + bta;
        const float mk = tmk[c], mv = tmv[c], mr = tmr[c];
        const size_t idx = (size_t)t * C_DIM + c;
        Ak[idx] = f2bf(xx * mk + sxs * (1.0f - mk));
        Av[idx] = f2bf(xx * mv + sxs * (1.0f - mv));
        Ar[idx] = f2bf(xx * mr + sxs * (1.0f - mr));
        if (t == T_DIM - 1) xx_last[c] = xx;
    }
}

// ---------------------------------------------------------------------------
// 3) bf16 WMMA GEMM:  D[M][N] = A[M][K] * Bt[N][K]^T  (f32 accumulate)
//    128x128 block tile, BK=64.  Waves in 4x2 grid; wave tile 32x64
//    -> 16 x v_wmma_f32_16x16x32_bf16 per stage, B fragments reused 2x.
//    Global->LDS staged with GLOBAL_LOAD_ASYNC_TO_LDS_B128 (ASYNCcnt) and
//    double-buffered LDS when available; sync fallback otherwise.
//    mode: 0 = store, 1 = sigmoid-store (r), 2 = add residual (final out)
// ---------------------------------------------------------------------------
__global__ __launch_bounds__(256)
void gemm_bf16_wmma_kernel(const unsigned short* __restrict__ A,
                           const unsigned short* __restrict__ Bt,
                           float* __restrict__ D,
                           const float* __restrict__ resid,
                           int M, int N, int K, int mode) {
#if USE_ASYNC_LDS
    __shared__ alignas(32) unsigned short Alds[2][BM][BK];   // 2 x 16 KB
    __shared__ alignas(32) unsigned short Blds[2][BN][BK];   // 2 x 16 KB
#else
    __shared__ alignas(32) unsigned short Alds[1][BM][BK];
    __shared__ alignas(32) unsigned short Blds[1][BN][BK];
#endif

    const int tid   = threadIdx.x;
    const int lane  = tid & 31;
    const int wave  = tid >> 5;          // 0..7
    const int waveM = wave & 3;          // 4 strips of 32 rows
    const int waveN = wave >> 2;         // 2 strips of 64 cols
    const int m0    = blockIdx.x * BM;
    const int n0    = blockIdx.y * BN;
    const int l16   = lane & 15;
    const int kh    = lane >> 4;         // wave32 half select

    v8f acc[2][4] = {};

    // stage loader: 1024 A chunks + 1024 B chunks of 16B, 4+4 per thread
    auto issueStage = [&](int buf, int k0) {
        #pragma unroll
        for (int i = 0; i < 4; i++) {
            const int chunk = tid + i * 256;
            const int r  = chunk >> 3;
            const int c8 = chunk & 7;
#if USE_ASYNC_LDS
            __builtin_amdgcn_global_load_async_to_lds_b128(
                (AS1 v4i_vs*)(A + (size_t)(m0 + r) * K + k0 + c8 * 8),
                (AS3 v4i_vs*)(&Alds[buf][r][c8 * 8]), 0, 0);
            __builtin_amdgcn_global_load_async_to_lds_b128(
                (AS1 v4i_vs*)(Bt + (size_t)(n0 + r) * K + k0 + c8 * 8),
                (AS3 v4i_vs*)(&Blds[buf][r][c8 * 8]), 0, 0);
#else
            *(v8u*)(&Alds[buf][r][c8 * 8]) =
                *(const v8u*)(A + (size_t)(m0 + r) * K + k0 + c8 * 8);
            *(v8u*)(&Blds[buf][r][c8 * 8]) =
                *(const v8u*)(Bt + (size_t)(n0 + r) * K + k0 + c8 * 8);
#endif
        }
    };

    auto computeStage = [&](int buf) {
        #pragma unroll
        for (int kk = 0; kk < BK; kk += 32) {
            // A fragments (16x32 bf16 per ISA layout):
            //  lanes 0-15:  elems 0..7 -> K 0..7,  elems 8..15 -> K 16..23
            //  lanes 16-31: elems 0..7 -> K 8..15, elems 8..15 -> K 24..31
            v16bf af[2];
            #pragma unroll
            for (int mt = 0; mt < 2; mt++) {
                const int arow = waveM * 32 + mt * 16 + l16;
                v8u alo = *(const v8u*)(&Alds[buf][arow][kk + kh * 8]);
                v8u ahi = *(const v8u*)(&Alds[buf][arow][kk + 16 + kh * 8]);
                v16u au = __builtin_shufflevector(alo, ahi,
                          0, 1, 2, 3, 4, 5, 6, 7, 8, 9, 10, 11, 12, 13, 14, 15);
                af[mt] = __builtin_bit_cast(v16bf, au);
            }
            // B fragments (32x16 bf16): same K layout as A fragments
            #pragma unroll
            for (int nt = 0; nt < 4; nt++) {
                const int bcol = waveN * 64 + nt * 16 + l16;
                v8u blo = *(const v8u*)(&Blds[buf][bcol][kk + kh * 8]);
                v8u bhi = *(const v8u*)(&Blds[buf][bcol][kk + 16 + kh * 8]);
                v16u bu = __builtin_shufflevector(blo, bhi,
                          0, 1, 2, 3, 4, 5, 6, 7, 8, 9, 10, 11, 12, 13, 14, 15);
                v16bf bfr = __builtin_bit_cast(v16bf, bu);
                acc[0][nt] = __builtin_amdgcn_wmma_f32_16x16x32_bf16(
                                 false, af[0], false, bfr, (short)0, acc[0][nt], false, false);
                acc[1][nt] = __builtin_amdgcn_wmma_f32_16x16x32_bf16(
                                 false, af[1], false, bfr, (short)0, acc[1][nt], false, false);
            }
        }
    };

    const int S = K / BK;
#if USE_ASYNC_LDS
    issueStage(0, 0);
    for (int s = 0; s < S; s++) {
        const int buf = s & 1;
        __builtin_amdgcn_s_wait_asynccnt(0);   // stage s data landed in LDS
        __syncthreads();                       // visible to all lanes
        if (s + 1 < S) issueStage(buf ^ 1, (s + 1) * BK);  // overlap with compute
        computeStage(buf);
        __syncthreads();                       // all reads of buf done
    }
#else
    for (int s = 0; s < S; s++) {
        issueStage(0, s * BK);
        __syncthreads();
        computeStage(0);
        __syncthreads();
    }
#endif

    // epilogue: C/D layout -> row = e + 8*kh, col = l16 within each 16x16 tile
    #pragma unroll
    for (int mt = 0; mt < 2; mt++) {
        #pragma unroll
        for (int nt = 0; nt < 4; nt++) {
            const int col = n0 + waveN * 64 + nt * 16 + l16;
            #pragma unroll
            for (int e = 0; e < 8; e++) {
                const int row = m0 + waveM * 32 + mt * 16 + kh * 8 + e;
                const size_t idx = (size_t)row * N + col;
                float d = acc[mt][nt][e];
                if (mode == 1)      d = 1.0f / (1.0f + __expf(-d));   // sigmoid(r)
                else if (mode == 2) d += resid[idx];                  // x + (r*wkv)@Wo
                D[idx] = d;
            }
        }
    }
}

// ---------------------------------------------------------------------------
// 4) WKV scan: sequential over T, parallel over C channels (f32, exact
//    reference recurrence); fuses rwkv = bf16(r * out).
// ---------------------------------------------------------------------------
__global__ __launch_bounds__(256)
void wkv_scan_kernel(const float* __restrict__ Kf, const float* __restrict__ Vf,
                     const float* __restrict__ Rf,
                     const float* __restrict__ aa0, const float* __restrict__ bb0,
                     const float* __restrict__ pp0,
                     const float* __restrict__ time_first,
                     const float* __restrict__ time_decay,
                     unsigned short* __restrict__ RW,
                     float* __restrict__ aa_out, float* __restrict__ bb_out,
                     float* __restrict__ pp_out) {
    const int c = blockIdx.x * blockDim.x + threadIdx.x;
    float aa = aa0[c], bb = bb0[c], pp = pp0[c];
    const float tf = time_first[c];
    const float w  = -__expf(time_decay[c]);
    for (int t = 0; t < T_DIM; t++) {
        const size_t idx = (size_t)t * C_DIM + c;
        const float kk = Kf[idx];
        const float vv = Vf[idx];
        const float ww = tf + kk;
        const float p  = fmaxf(pp, ww);
        const float e1 = __expf(pp - p);
        const float e2 = __expf(ww - p);
        const float out = (e1 * aa + e2 * vv) / (e1 * bb + e2);
        const float ww2 = w + pp;
        const float p2  = fmaxf(ww2, kk);
        const float e1b = __expf(ww2 - p2);
        const float e2b = __expf(kk - p2);
        aa = e1b * aa + e2b * vv;
        bb = e1b * bb + e2b;
        pp = p2;
        RW[idx] = f2bf(Rf[idx] * out);
    }
    aa_out[c] = aa; bb_out[c] = bb; pp_out[c] = pp;
}

// ---------------------------------------------------------------------------
// Host launcher
// ---------------------------------------------------------------------------
extern "C" void kernel_launch(void* const* d_in, const int* in_sizes, int n_in,
                              void* d_out, int out_size, void* d_ws, size_t ws_size,
                              hipStream_t stream) {
    const float* x   = (const float*)d_in[0];
    const float* sx  = (const float*)d_in[1];
    const float* aa0 = (const float*)d_in[2];
    const float* bb0 = (const float*)d_in[3];
    const float* pp0 = (const float*)d_in[4];
    const float* tf  = (const float*)d_in[5];
    const float* td  = (const float*)d_in[6];
    const float* tmk = (const float*)d_in[7];
    const float* tmv = (const float*)d_in[8];
    const float* tmr = (const float*)d_in[9];
    const float* lns = (const float*)d_in[10];
    const float* lnb = (const float*)d_in[11];
    const float* Wk  = (const float*)d_in[12];
    const float* Wv  = (const float*)d_in[13];
    const float* Wr  = (const float*)d_in[14];
    const float* Wo  = (const float*)d_in[15];

    // outputs: out[T*C] | xx_last[C] | aa[C] | bb[C] | pp[C]
    float* out     = (float*)d_out;
    float* xx_last = out + (size_t)T_DIM * C_DIM;
    float* aa_out  = xx_last + C_DIM;
    float* bb_out  = aa_out + C_DIM;
    float* pp_out  = bb_out + C_DIM;

    // workspace layout
    char* p = (char*)d_ws;
    const size_t actBF = (size_t)T_DIM * C_DIM * sizeof(unsigned short); // 32 MB
    const size_t wBF   = (size_t)C_DIM * C_DIM * sizeof(unsigned short); //  8 MB
    const size_t actF  = (size_t)T_DIM * C_DIM * sizeof(float);          // 64 MB
    unsigned short* Ak  = (unsigned short*)p;            p += actBF;
    unsigned short* Av  = (unsigned short*)p;            p += actBF;
    unsigned short* Ar  = (unsigned short*)p;            p += actBF;
    unsigned short* WkT = (unsigned short*)p;            p += wBF;
    unsigned short* WvT = (unsigned short*)p;            p += wBF;
    unsigned short* WrT = (unsigned short*)p;            p += wBF;
    unsigned short* WoT = (unsigned short*)p;            p += wBF;
    float*          Kf  = (float*)p;                     p += actF;
    float*          Vf  = (float*)p;                     p += actF;
    float*          Rf  = (float*)p;                     p += actF;
    unsigned short* RW  = (unsigned short*)p;            p += actBF;

    // 1) weight transpose + bf16
    {
        dim3 grid(C_DIM / 32, C_DIM / 32), blk(32, 8);
        transpose_bf16_kernel<<<grid, blk, 0, stream>>>(Wk, WkT);
        transpose_bf16_kernel<<<grid, blk, 0, stream>>>(Wv, WvT);
        transpose_bf16_kernel<<<grid, blk, 0, stream>>>(Wr, WrT);
        transpose_bf16_kernel<<<grid, blk, 0, stream>>>(Wo, WoT);
    }

    // 2) LN + mixing
    ln_mix_kernel<<<T_DIM, 256, 0, stream>>>(x, sx, tmk, tmv, tmr, lns, lnb,
                                             Ak, Av, Ar, xx_last);

    // 3) k / v / r GEMMs
    {
        dim3 grid(T_DIM / BM, C_DIM / BN), blk(256);
        gemm_bf16_wmma_kernel<<<grid, blk, 0, stream>>>(Ak, WkT, Kf, nullptr,
                                                        T_DIM, C_DIM, C_DIM, 0);
        gemm_bf16_wmma_kernel<<<grid, blk, 0, stream>>>(Av, WvT, Vf, nullptr,
                                                        T_DIM, C_DIM, C_DIM, 0);
        gemm_bf16_wmma_kernel<<<grid, blk, 0, stream>>>(Ar, WrT, Rf, nullptr,
                                                        T_DIM, C_DIM, C_DIM, 1);
    }

    // 4) WKV scan (+ fuse r*wkv -> bf16)
    wkv_scan_kernel<<<C_DIM / 256, 256, 0, stream>>>(Kf, Vf, Rf, aa0, bb0, pp0,
                                                     tf, td, RW,
                                                     aa_out, bb_out, pp_out);

    // 5) out = x + (r*wkv) @ Wo
    {
        dim3 grid(T_DIM / BM, C_DIM / BN), blk(256);
        gemm_bf16_wmma_kernel<<<grid, blk, 0, stream>>>(RW, WoT, out, x,
                                                        T_DIM, C_DIM, C_DIM, 2);
    }
}
